// GumbelSoftmaxTokenizer_8907762172284
// MI455X (gfx1250) — compile-verified, matrix operand order checked by reading, not run on hardware
//
#include <hip/hip_runtime.h>

typedef __attribute__((ext_vector_type(16))) _Float16 half16;
typedef __attribute__((ext_vector_type(8)))  _Float16 half8;
typedef __attribute__((ext_vector_type(8)))  float    float8;
typedef int int4v __attribute__((__vector_size__(16)));

#define NB   8
#define PN   16384
#define MT   128
#define KNN  16
#define TOKD 768
#define IH   256

#define GAS __attribute__((address_space(1)))
#define LAS __attribute__((address_space(3)))

#if defined(__gfx1250__) && __has_builtin(__builtin_amdgcn_global_load_async_to_lds_b128) && __has_builtin(__builtin_amdgcn_s_wait_asynccnt)
#define USE_ASYNC 1
#else
#define USE_ASYNC 0
#endif

// ---------------------------------------------------------------------------
// Weight convert: W (K x N, f32, row-major) -> Wt (N x Kpad, f16), zero-padded
// ---------------------------------------------------------------------------
__global__ void convert_wt(const float* __restrict__ W, _Float16* __restrict__ Wt,
                           int K, int N, int Kpad) {
  int i = blockIdx.x * blockDim.x + threadIdx.x;
  if (i >= N * Kpad) return;
  int n = i / Kpad, k = i - n * Kpad;
  Wt[(size_t)n * Kpad + k] = (k < K) ? (_Float16)W[(size_t)k * N + n] : (_Float16)0.f;
}

// features (rows x 6 f32) -> A0 (rows x 32 f16, zero padded)
__global__ void pad_features(const float* __restrict__ f, _Float16* __restrict__ A0, int rows) {
  int i = blockIdx.x * blockDim.x + threadIdx.x;
  if (i >= rows * 32) return;
  int r = i >> 5, c = i & 31;
  A0[i] = (c < 6) ? (_Float16)f[(size_t)r * 6 + c] : (_Float16)0.f;
}

// fill zcat cols 768..799: 4 coords then zeros
__global__ void zcat_tail(const float* __restrict__ coords, _Float16* __restrict__ zcat, int rows) {
  int i = blockIdx.x * blockDim.x + threadIdx.x;
  if (i >= rows * 32) return;
  int r = i >> 5, c = i & 31;
  _Float16 v = (_Float16)0.f;
  if (c < 4) v = (_Float16)coords[(size_t)r * 5 + 1 + c];
  zcat[(size_t)r * 800 + 768 + c] = v;
}

// ---------------------------------------------------------------------------
// WMMA GEMM: C[M x N] = act(A[M x K](f16) @ Wt^T + bias), Wt stored [N][ldw]
// tile 128x128x32, 256 threads = 8 waves, each wave does a 32x64 patch.
// Async double-buffered LDS staging (GLOBAL_LOAD_ASYNC_TO_LDS_B128) when
// available; sync global->VGPR->LDS fallback otherwise.
// ---------------------------------------------------------------------------
#define LDT 40   // LDS row stride in halves (80B, keeps 16B alignment)

template<bool RELU, bool OUT16>
__global__ __launch_bounds__(256)
void gemm_wmma(const _Float16* __restrict__ A, int lda,
               const _Float16* __restrict__ Wt, int ldw,
               const float* __restrict__ bias,
               void* __restrict__ Cout, int ldc, int Ktot) {
#if USE_ASYNC
  __shared__ _Float16 As[2][128 * LDT];
  __shared__ _Float16 Bs[2][128 * LDT];
#else
  __shared__ _Float16 As[1][128 * LDT];
  __shared__ _Float16 Bs[1][128 * LDT];
#endif
  const int tid  = threadIdx.x;
  const int lane = tid & 31;
  const int wave = tid >> 5;
  const int wm   = wave >> 1;          // 0..3 (M)
  const int wn   = wave & 1;           // 0..1 (N)
  const int m0   = blockIdx.x * 128;
  const int n0   = blockIdx.y * 128;

  float8 acc[2][4];
#pragma unroll
  for (int tm = 0; tm < 2; ++tm)
#pragma unroll
    for (int tn = 0; tn < 4; ++tn)
#pragma unroll
      for (int r = 0; r < 8; ++r) acc[tm][tn][r] = 0.f;

  // stage one 128x32 A tile + 128x32 Wt tile into LDS buffer `buf`
  auto stage = [&](int buf, int k0) {
#pragma unroll
    for (int it = 0; it < 2; ++it) {
      int lin = (it * 256 + tid) * 8;      // half index in tile
      int r = lin >> 5, c = lin & 31;
      const _Float16* ga = &A [(size_t)(m0 + r) * lda + k0 + c];
      const _Float16* gb = &Wt[(size_t)(n0 + r) * ldw + k0 + c];
#if USE_ASYNC
      __builtin_amdgcn_global_load_async_to_lds_b128(
          (GAS int4v*)(uintptr_t)ga,
          (LAS int4v*)(unsigned)(uintptr_t)&As[buf][r * LDT + c], 0, 0);
      __builtin_amdgcn_global_load_async_to_lds_b128(
          (GAS int4v*)(uintptr_t)gb,
          (LAS int4v*)(unsigned)(uintptr_t)&Bs[buf][r * LDT + c], 0, 0);
#else
      *(half8*)&As[buf][r * LDT + c] = *(const half8*)ga;
      *(half8*)&Bs[buf][r * LDT + c] = *(const half8*)gb;
#endif
    }
  };

  // fragments + 8 WMMAs from LDS buffer `buf`
  // 16-bit A-matrix layout: lanes 0-15 hold K 0-7/16-23, lanes 16-31 hold K 8-15/24-31
  auto compute = [&](int buf) {
    const int kb = (lane >> 4) * 8;
    half16 af[2], bf[4];
#pragma unroll
    for (int tm = 0; tm < 2; ++tm) {
      int mr = wm * 32 + tm * 16 + (lane & 15);
      half8 lo = *(const half8*)&As[buf][mr * LDT + kb];
      half8 hi = *(const half8*)&As[buf][mr * LDT + 16 + kb];
#pragma unroll
      for (int i = 0; i < 8; ++i) { af[tm][i] = lo[i]; af[tm][i + 8] = hi[i]; }
    }
#pragma unroll
    for (int tn = 0; tn < 4; ++tn) {
      int nr = wn * 64 + tn * 16 + (lane & 15);
      half8 lo = *(const half8*)&Bs[buf][nr * LDT + kb];
      half8 hi = *(const half8*)&Bs[buf][nr * LDT + 16 + kb];
#pragma unroll
      for (int i = 0; i < 8; ++i) { bf[tn][i] = lo[i]; bf[tn][i + 8] = hi[i]; }
    }
#pragma unroll
    for (int tm = 0; tm < 2; ++tm)
#pragma unroll
      for (int tn = 0; tn < 4; ++tn)
        acc[tm][tn] = __builtin_amdgcn_wmma_f32_16x16x32_f16(
            false, af[tm], false, bf[tn], (short)0, acc[tm][tn], false, false);
  };

#if USE_ASYNC
  stage(0, 0);
  __builtin_amdgcn_s_wait_asynccnt(0);
  __syncthreads();
  int cur = 0;
  for (int k0 = 0; k0 < Ktot; k0 += 32) {
    int nxt = cur ^ 1;
    if (k0 + 32 < Ktot) stage(nxt, k0 + 32);   // overlap next tile with compute
    compute(cur);
    __builtin_amdgcn_s_wait_asynccnt(0);
    __syncthreads();
    cur = nxt;
  }
#else
  for (int k0 = 0; k0 < Ktot; k0 += 32) {
    stage(0, k0);
    if (k0 + 32 < Ktot) {
      __builtin_prefetch(&A [(size_t)(m0 + (tid >> 1)) * lda + k0 + 32], 0, 1);
      __builtin_prefetch(&Wt[(size_t)(n0 + (tid >> 1)) * ldw + k0 + 32], 0, 1);
    }
    __syncthreads();
    compute(0);
    __syncthreads();
  }
#endif

  // epilogue: C 16x16 layout (lanes 0-15 -> M r, lanes 16-31 -> M 8+r)
#pragma unroll
  for (int tm = 0; tm < 2; ++tm) {
#pragma unroll
    for (int tn = 0; tn < 4; ++tn) {
      int col = n0 + wn * 64 + tn * 16 + (lane & 15);
      float bv = bias[col];
#pragma unroll
      for (int r = 0; r < 8; ++r) {
        int row = m0 + wm * 32 + tm * 16 + (lane >> 4) * 8 + r;
        float v = acc[tm][tn][r] + bv;
        if (RELU) v = v > 0.f ? v : 0.f;
        if (OUT16) ((_Float16*)Cout)[(size_t)row * ldc + col] = (_Float16)v;
        else       ((float*)    Cout)[(size_t)row * ldc + col] = v;
      }
    }
  }
}

// ---------------------------------------------------------------------------
// LayerNorm over 256 cols, one wave32 per row (8 halves per lane)
// ---------------------------------------------------------------------------
__global__ __launch_bounds__(256)
void ln_kernel(const _Float16* __restrict__ z, _Float16* __restrict__ o,
               const float* __restrict__ g, const float* __restrict__ bta, int rows) {
  int wave = threadIdx.x >> 5, lane = threadIdx.x & 31;
  int row = blockIdx.x * 8 + wave;
  if (row >= rows) return;
  const _Float16* src = z + (size_t)row * IH;
  float v[8], s = 0.f;
#pragma unroll
  for (int i = 0; i < 8; ++i) { v[i] = (float)src[lane * 8 + i]; s += v[i]; }
#pragma unroll
  for (int off = 16; off > 0; off >>= 1) s += __shfl_xor(s, off, 32);
  float mu = s * (1.f / IH);
  float q = 0.f;
#pragma unroll
  for (int i = 0; i < 8; ++i) { float d = v[i] - mu; q += d * d; }
#pragma unroll
  for (int off = 16; off > 0; off >>= 1) q += __shfl_xor(q, off, 32);
  float inv = rsqrtf(q * (1.f / IH) + 1e-5f);
  _Float16* dst = o + (size_t)row * IH;
#pragma unroll
  for (int i = 0; i < 8; ++i) {
    int c = lane * 8 + i;
    dst[c] = (_Float16)((v[i] - mu) * inv * g[c] + bta[c]);
  }
}

// importance dot + gumbel perturb, one wave per row
__global__ __launch_bounds__(256)
void imp_kernel(const _Float16* __restrict__ z2, const float* __restrict__ iw3,
                const float* __restrict__ ib3, const float* __restrict__ noise,
                const float* __restrict__ lt, float* __restrict__ pert, int rows) {
  int wave = threadIdx.x >> 5, lane = threadIdx.x & 31;
  int row = blockIdx.x * 8 + wave;
  if (row >= rows) return;
  const _Float16* src = z2 + (size_t)row * IH;
  float s = 0.f;
#pragma unroll
  for (int i = 0; i < 8; ++i) { int c = lane * 8 + i; s += (float)src[c] * iw3[c]; }
#pragma unroll
  for (int off = 16; off > 0; off >>= 1) s += __shfl_xor(s, off, 32);
  if (lane == 0) {
    float t = expf(lt[0]); t = t > 0.1f ? t : 0.1f;
    pert[row] = (s + ib3[0] + noise[row]) / t;
  }
}

__device__ __forceinline__ bool argmax_gt(float a, int ia, float b, int ib) {
  return (a > b) || (a == b && ia < ib);
}
__device__ __forceinline__ bool argmin_lt(float a, int ia, float b, int ib) {
  return (a < b) || (a == b && ia < ib);
}

// per-batch iterative top-128 (descending, ties by lower index)
__global__ __launch_bounds__(256)
void topk_kernel(const float* __restrict__ pert, float* __restrict__ pertc,
                 int* __restrict__ sel, float* __restrict__ cent,
                 const float* __restrict__ coords) {
  int b = blockIdx.x, tid = threadIdx.x;
  for (int i = tid; i < PN; i += 256) pertc[b * PN + i] = pert[b * PN + i];
  __syncthreads();
  __shared__ float sv[256];
  __shared__ int   si[256];
  for (int k = 0; k < MT; ++k) {
    float bv = -3.0e38f; int bi = 0x7fffffff;
    for (int i = tid; i < PN; i += 256) {
      float v = pertc[b * PN + i];
      if (argmax_gt(v, i, bv, bi)) { bv = v; bi = i; }
    }
    sv[tid] = bv; si[tid] = bi;
    __syncthreads();
    for (int s = 128; s > 0; s >>= 1) {
      if (tid < s && argmax_gt(sv[tid + s], si[tid + s], sv[tid], si[tid])) {
        sv[tid] = sv[tid + s]; si[tid] = si[tid + s];
      }
      __syncthreads();
    }
    if (tid == 0) {
      int idx = si[0];
      sel[b * MT + k] = idx;
      pertc[b * PN + idx] = -3.0e38f;
      for (int c = 0; c < 4; ++c)
        cent[((size_t)b * MT + k) * 4 + c] = coords[((size_t)b * PN + idx) * 5 + 1 + c];
    }
    __syncthreads();
  }
}

// kNN (top-16 smallest d2) + max-pool of pf features; one block per centroid
__global__ __launch_bounds__(256)
void knn_pool_kernel(const float* __restrict__ cent, const float* __restrict__ coords,
                     const _Float16* __restrict__ zcat, _Float16* __restrict__ pooled) {
  int ci = blockIdx.x, b = blockIdx.y, tid = threadIdx.x;
  int g = b * MT + ci;
  float cx = cent[(size_t)g * 4 + 0], cy = cent[(size_t)g * 4 + 1];
  float cz = cent[(size_t)g * 4 + 2], ct = cent[(size_t)g * 4 + 3];

  float d[KNN]; int id[KNN];
#pragma unroll
  for (int j = 0; j < KNN; ++j) { d[j] = 3.0e38f; id[j] = 0x7fffffff; }

  for (int p = tid; p < PN; p += 256) {
    const float* pc = &coords[((size_t)b * PN + p) * 5 + 1];
    float dx = cx - pc[0], dy = cy - pc[1], dz = cz - pc[2], dt = ct - pc[3];
    float d2 = dx * dx + dy * dy + dz * dz + dt * dt;
    if (argmin_lt(d2, p, d[KNN - 1], id[KNN - 1])) {
      d[KNN - 1] = d2; id[KNN - 1] = p;
#pragma unroll
      for (int j = KNN - 1; j > 0; --j) {
        if (argmin_lt(d[j], id[j], d[j - 1], id[j - 1])) {
          float td = d[j]; d[j] = d[j - 1]; d[j - 1] = td;
          int   ti = id[j]; id[j] = id[j - 1]; id[j - 1] = ti;
        }
      }
    }
  }

  __shared__ float cd[256 * KNN];
  __shared__ int   cix[256 * KNN];
  __shared__ float rv[256];
  __shared__ int   ri[256];
  __shared__ int   knnIdx[KNN];
#pragma unroll
  for (int j = 0; j < KNN; ++j) { cd[tid * KNN + j] = d[j]; cix[tid * KNN + j] = id[j]; }
  __syncthreads();

  int head = 0;
  for (int k = 0; k < KNN; ++k) {
    float v = (head < KNN) ? cd[tid * KNN + head] : 3.0e38f;
    int   ix = (head < KNN) ? cix[tid * KNN + head] : 0x7fffffff;
    rv[tid] = v; ri[tid] = ix;
    __syncthreads();
    for (int s = 128; s > 0; s >>= 1) {
      if (tid < s && argmin_lt(rv[tid + s], ri[tid + s], rv[tid], ri[tid])) {
        rv[tid] = rv[tid + s]; ri[tid] = ri[tid + s];
      }
      __syncthreads();
    }
    if (tid == 0) knnIdx[k] = ri[0];
    __syncthreads();
    if (head < KNN && cix[tid * KNN + head] == knnIdx[k]) head++;
    __syncthreads();
  }

  // max-pool pf over the 16 neighbors (pf lives in zcat cols 0..767, stride 800)
  for (int c = tid; c < TOKD; c += 256) {
    float m = -3.0e38f;
#pragma unroll
    for (int j = 0; j < KNN; ++j) {
      int p = knnIdx[j];
      float v = (float)zcat[((size_t)b * PN + p) * 800 + c];
      m = v > m ? v : m;
    }
    pooled[(size_t)g * TOKD + c] = (_Float16)m;
  }
}

// stable argsort by cent t, scatter tokens/cents/masks into d_out
__global__ __launch_bounds__(128)
void order_scatter(const float* __restrict__ tok, const float* __restrict__ cent,
                   float* __restrict__ out) {
  int b = blockIdx.x, i = threadIdx.x;
  __shared__ float keys[MT];
  keys[i] = cent[((size_t)b * MT + i) * 4 + 3];
  __syncthreads();
  float ki = keys[i];
  int rank = 0;
  for (int j = 0; j < MT; ++j) {
    float kj = keys[j];
    rank += (kj < ki) || (kj == ki && j < i);
  }
  const float* src = tok + ((size_t)b * MT + i) * TOKD;
  float* dst = out + ((size_t)b * MT + rank) * TOKD;
  for (int c = 0; c < TOKD; ++c) dst[c] = src[c];
  float* ocent = out + (size_t)NB * MT * TOKD;
  for (int c = 0; c < 4; ++c)
    ocent[((size_t)b * MT + rank) * 4 + c] = cent[((size_t)b * MT + i) * 4 + c];
  float* om = out + (size_t)NB * MT * TOKD + (size_t)NB * MT * 4;
  om[b * MT + i] = 1.0f;
}

// ---------------------------------------------------------------------------
extern "C" void kernel_launch(void* const* d_in, const int* in_sizes, int n_in,
                              void* d_out, int out_size, void* d_ws, size_t ws_size,
                              hipStream_t stream) {
  (void)n_in; (void)out_size; (void)ws_size;
  const float* coords = (const float*)d_in[0];
  const float* feats  = (const float*)d_in[1];
  const float* ltemp  = (const float*)d_in[2];
  const float* w1 = (const float*)d_in[3];  const float* b1 = (const float*)d_in[4];
  const float* w2 = (const float*)d_in[5];  const float* b2 = (const float*)d_in[6];
  const float* w3 = (const float*)d_in[7];  const float* b3 = (const float*)d_in[8];
  const float* w4 = (const float*)d_in[9];  const float* b4 = (const float*)d_in[10];
  const float* iw1 = (const float*)d_in[11]; const float* ib1 = (const float*)d_in[12];
  const float* lng = (const float*)d_in[13]; const float* lnb = (const float*)d_in[14];
  const float* iw2 = (const float*)d_in[15]; const float* ib2 = (const float*)d_in[16];
  const float* iw3 = (const float*)d_in[17]; const float* ib3 = (const float*)d_in[18];
  const float* nw1 = (const float*)d_in[19]; const float* nb1 = (const float*)d_in[20];
  const float* nw2 = (const float*)d_in[21]; const float* nb2 = (const float*)d_in[22];
  const float* gumbel = (const float*)d_in[23];

  const int rows = in_sizes[1] / 6;          // B*P = 131072

  char* ws = (char*)d_ws;
  size_t off = 0;
  auto wsAlloc = [&](size_t bytes) -> char* {
    off = (off + 255) & ~(size_t)255;
    char* p = ws + off; off += bytes; return p;
  };
  _Float16* w1t  = (_Float16*)wsAlloc((size_t)256 * 32 * 2);
  _Float16* w2t  = (_Float16*)wsAlloc((size_t)512 * 256 * 2);
  _Float16* w3t  = (_Float16*)wsAlloc((size_t)768 * 512 * 2);
  _Float16* w4t  = (_Float16*)wsAlloc((size_t)768 * 768 * 2);
  _Float16* iw1t = (_Float16*)wsAlloc((size_t)256 * 800 * 2);
  _Float16* iw2t = (_Float16*)wsAlloc((size_t)256 * 256 * 2);
  _Float16* nw1t = (_Float16*)wsAlloc((size_t)768 * 768 * 2);
  _Float16* nw2t = (_Float16*)wsAlloc((size_t)768 * 768 * 2);
  _Float16* A0   = (_Float16*)wsAlloc((size_t)rows * 32 * 2);
  _Float16* bufA = (_Float16*)wsAlloc((size_t)rows * 256 * 2); // h1, then z1ln
  _Float16* bufB = (_Float16*)wsAlloc((size_t)rows * 512 * 2); // h2, then z2
  _Float16* bufC = (_Float16*)wsAlloc((size_t)rows * 768 * 2); // h3, then z1
  _Float16* zcat = (_Float16*)wsAlloc((size_t)rows * 800 * 2); // pf | coords4 | 0-pad
  float* pert  = (float*)wsAlloc((size_t)rows * 4);
  float* pertc = (float*)wsAlloc((size_t)rows * 4);
  int*   sel   = (int*)wsAlloc((size_t)NB * MT * 4);
  float* cent  = (float*)wsAlloc((size_t)NB * MT * 4 * 4);
  _Float16* pooled = (_Float16*)wsAlloc((size_t)NB * MT * TOKD * 2);
  _Float16* t1     = (_Float16*)wsAlloc((size_t)NB * MT * TOKD * 2);
  float* tokw = (float*)wsAlloc((size_t)NB * MT * TOKD * 4);

  auto cvt = [&](const float* W, _Float16* Wt, int K, int N, int Kpad) {
    int tot = N * Kpad;
    convert_wt<<<(tot + 255) / 256, 256, 0, stream>>>(W, Wt, K, N, Kpad);
  };
  cvt(w1,  w1t,    6, 256,  32);
  cvt(w2,  w2t,  256, 512, 256);
  cvt(w3,  w3t,  512, 768, 512);
  cvt(w4,  w4t,  768, 768, 768);
  cvt(iw1, iw1t, 772, 256, 800);
  cvt(iw2, iw2t, 256, 256, 256);
  cvt(nw1, nw1t, 768, 768, 768);
  cvt(nw2, nw2t, 768, 768, 768);

  pad_features<<<(rows * 32 + 255) / 256, 256, 0, stream>>>(feats, A0, rows);

  const int MB = rows / 128;  // 1024 row-blocks
  // h1 = relu(A0 @ w1 + b1)
  gemm_wmma<true, true><<<dim3(MB, 2), 256, 0, stream>>>(A0, 32, w1t, 32, b1, bufA, 256, 32);
  // h2 = relu(h1 @ w2 + b2)
  gemm_wmma<true, true><<<dim3(MB, 4), 256, 0, stream>>>(bufA, 256, w2t, 256, b2, bufB, 512, 256);
  // h3 = relu(h2 @ w3 + b3)
  gemm_wmma<true, true><<<dim3(MB, 6), 256, 0, stream>>>(bufB, 512, w3t, 512, b3, bufC, 768, 512);
  // pf = h3 @ w4 + b4  -> zcat cols 0..767 (stride 800)
  gemm_wmma<false, true><<<dim3(MB, 6), 256, 0, stream>>>(bufC, 768, w4t, 768, b4, zcat, 800, 768);
  // zcat cols 768..799 = coords4 | zeros
  zcat_tail<<<(rows * 32 + 255) / 256, 256, 0, stream>>>(coords, zcat, rows);
  // z1 = relu(zcat @ iw1 + ib1)
  gemm_wmma<true, true><<<dim3(MB, 2), 256, 0, stream>>>(zcat, 800, iw1t, 800, ib1, bufC, 256, 800);
  // z1ln = LN(z1)
  ln_kernel<<<rows / 8, 256, 0, stream>>>(bufC, bufA, lng, lnb, rows);
  // z2 = relu(z1ln @ iw2 + ib2)
  gemm_wmma<true, true><<<dim3(MB, 2), 256, 0, stream>>>(bufA, 256, iw2t, 256, ib2, bufB, 256, 256);
  // pert = (z2 @ iw3 + ib3 + noise) / temp
  imp_kernel<<<rows / 8, 256, 0, stream>>>(bufB, iw3, ib3, gumbel, ltemp, pert, rows);
  // per-batch top-128 + centroid gather
  topk_kernel<<<NB, 256, 0, stream>>>(pert, pertc, sel, cent, coords);
  // kNN + max-pool
  knn_pool_kernel<<<dim3(MT, NB), 256, 0, stream>>>(cent, coords, zcat, pooled);
  // t1 = relu(pooled @ nw1 + nb1); tok = t1 @ nw2 + nb2
  gemm_wmma<true,  true ><<<dim3(NB, 6), 256, 0, stream>>>(pooled, 768, nw1t, 768, nb1, t1, 768, 768);
  gemm_wmma<false, false><<<dim3(NB, 6), 256, 0, stream>>>(t1, 768, nw2t, 768, nb2, tokw, 768, 768);
  // argsort by t + scatter tokens/cents/masks
  order_scatter<<<NB, MT, 0, stream>>>(tokw, cent, (float*)d_out);
}
